// SelfAttention_44444321579154
// MI455X (gfx1250) — matrix operand order repaired and checked by reading, not compiled, not run on hardware
//
#include <hip/hip_runtime.h>

// ---------------------------------------------------------------------------
// Types for CDNA5 WMMA (gfx1250, wave32)
// ---------------------------------------------------------------------------
typedef __attribute__((ext_vector_type(16))) __bf16 v16bf;
typedef __attribute__((ext_vector_type(8)))  __bf16 v8bf;
typedef __attribute__((ext_vector_type(4)))  __bf16 v4bf;
typedef __attribute__((ext_vector_type(8)))  float  v8f;

// Native converts: let the backend pick v_cvt_*_bf16 forms (RNE, matches ref)
__device__ __forceinline__ __bf16 f2bf(float f) { return static_cast<__bf16>(f); }
__device__ __forceinline__ float  bf2f(__bf16 b) { return static_cast<float>(b); }

__device__ __forceinline__ v8f wmma_bf16(v16bf a, v16bf b, v8f c) {
    // D = A(16x32 bf16) * B(32x16 bf16) + C(16x16 f32)
    return __builtin_amdgcn_wmma_f32_16x16x32_bf16(
        /*neg_a=*/false, a, /*neg_b=*/false, b,
        /*c_mod=*/(short)0, c, /*reuse_a=*/false, /*reuse_b=*/false);
}

// ---------------------------------------------------------------------------
// Constants
// ---------------------------------------------------------------------------
#define BATCH   64
#define SEQ     128
#define HID     512
#define NH      8
#define HD      64
#define ROWS    (BATCH*SEQ)          // 8192

// B-fragment packing: within a matrix [K,N], fragment (ks, nt) lane l elem e:
//   K = ks*32 + (e>=8)*16 + (l>=16)*8 + (e&7),  N = nt*16 + (l&15)
#define WBIG_ELEMS   (16*32*32*16)   // 512x512 = 262144 bf16 per matrix
#define WSMALL_ELEMS (2*4*32*16)     // 64x64   = 4096   bf16 per matrix

// ---------------------------------------------------------------------------
// Kernel 0: repack f32 weights into bf16 WMMA B-fragment layout in workspace
// ---------------------------------------------------------------------------
__global__ __launch_bounds__(256) void repack_weights(
    const float* __restrict__ Wq, const float* __restrict__ Wk,
    const float* __restrict__ Wv, const float* __restrict__ Wsq,
    const float* __restrict__ Wsk, const float* __restrict__ Wsv,
    __bf16* __restrict__ wbig, __bf16* __restrict__ wsmall) {
    int idx = blockIdx.x * 256 + threadIdx.x;
    const int BIG = 3 * WBIG_ELEMS;                   // 786432
    if (idx < BIG) {
        int mat = idx / WBIG_ELEMS, r = idx % WBIG_ELEMS;
        int e = r & 15, lane = (r >> 4) & 31, t = r >> 9;
        int nt = t & 31, ks = t >> 5;                 // ks in [0,16)
        int n = nt * 16 + (lane & 15);
        int k = ks * 32 + ((e >> 3) << 4) + ((lane >> 4) << 3) + (e & 7);
        const float* W = (mat == 0) ? Wq : (mat == 1) ? Wk : Wv;
        wbig[idx] = f2bf(W[k * HID + n]);
    } else {
        int id2 = idx - BIG;
        if (id2 < 3 * WSMALL_ELEMS) {
            int mat = id2 / WSMALL_ELEMS, r = id2 % WSMALL_ELEMS;
            int e = r & 15, lane = (r >> 4) & 31, t = r >> 9;
            int nt = t & 3, ks = t >> 2;              // ks in {0,1}
            int n = nt * 16 + (lane & 15);
            int k = ks * 32 + ((e >> 3) << 4) + ((lane >> 4) << 3) + (e & 7);
            const float* W = (mat == 0) ? Wsq : (mat == 1) ? Wsk : Wsv;
            wsmall[id2] = f2bf(W[k * HD + n]);
        }
    }
}

// ---------------------------------------------------------------------------
// Kernel 1: q,k,v = hidden @ W{q,k,v} + b   via bf16 WMMA
// grid (64, 8): block owns 128 M-rows x 64 N-cols for all three matrices.
// 8 waves, wave w owns M-strip [16w,16w+16); 12 v8f accumulators per wave.
// A-fragment f32 data is software-pipelined one k-step ahead.
// ---------------------------------------------------------------------------
__global__ __launch_bounds__(256) void qkv_kernel(
    const float* __restrict__ hidden, const __bf16* __restrict__ wbig,
    const float* __restrict__ bq, const float* __restrict__ bk,
    const float* __restrict__ bv, float* __restrict__ qkv) {
    const int mblk = blockIdx.x;           // 0..63
    const int nblk = blockIdx.y;           // 0..7
    const int tid  = threadIdx.x;
    const int w    = tid >> 5;
    const int lane = tid & 31;
    const int hk   = lane >> 4;            // K-half select for A layout
    const int ln   = lane & 15;

    v8f acc[3][4];
    #pragma unroll
    for (int m = 0; m < 3; m++)
        #pragma unroll
        for (int n = 0; n < 4; n++) { v8f z = {}; acc[m][n] = z; }

    const int rowg = mblk * 128 + w * 16 + ln;   // this lane's A-row (M)
    const float* arowbase = hidden + (size_t)rowg * HID + hk * 8;

    // prologue: prefetch k-step 0 A data
    float4 nx0 = *(const float4*)(arowbase + 0);
    float4 nx1 = *(const float4*)(arowbase + 4);
    float4 nx2 = *(const float4*)(arowbase + 16);
    float4 nx3 = *(const float4*)(arowbase + 20);

    for (int ks = 0; ks < 16; ks++) {
        float4 c0 = nx0, c1 = nx1, c2 = nx2, c3 = nx3;
        if (ks < 15) {                     // issue next k-step loads early
            const float* an = arowbase + (ks + 1) * 32;
            nx0 = *(const float4*)(an + 0);
            nx1 = *(const float4*)(an + 4);
            nx2 = *(const float4*)(an + 16);
            nx3 = *(const float4*)(an + 20);
        }
        // --- A fragment: 16x32 bf16; elem e: K = hk*8+e (e<8) / 16+hk*8+(e-8)
        union { v16bf v; __bf16 e[16]; } a;
        a.e[0]  = f2bf(c0.x); a.e[1]  = f2bf(c0.y); a.e[2]  = f2bf(c0.z); a.e[3]  = f2bf(c0.w);
        a.e[4]  = f2bf(c1.x); a.e[5]  = f2bf(c1.y); a.e[6]  = f2bf(c1.z); a.e[7]  = f2bf(c1.w);
        a.e[8]  = f2bf(c2.x); a.e[9]  = f2bf(c2.y); a.e[10] = f2bf(c2.z); a.e[11] = f2bf(c2.w);
        a.e[12] = f2bf(c3.x); a.e[13] = f2bf(c3.y); a.e[14] = f2bf(c3.z); a.e[15] = f2bf(c3.w);
        #pragma unroll
        for (int mat = 0; mat < 3; mat++) {
            const __bf16* wb = wbig + mat * WBIG_ELEMS;
            #pragma unroll
            for (int nt = 0; nt < 4; nt++) {
                int ntg = nblk * 4 + nt;
                v16bf bf = *(const v16bf*)(wb + (((ks * 32 + ntg) * 32) + lane) * 16);
                acc[mat][nt] = wmma_bf16(a.v, bf, acc[mat][nt]);
            }
        }
    }
    // --- store + bias. C layout: VGPR r -> row = hk*8 + r, col = ln (within tile)
    #pragma unroll
    for (int mat = 0; mat < 3; mat++) {
        const float* bias = (mat == 0) ? bq : (mat == 1) ? bk : bv;
        float* dst = qkv + (size_t)mat * ROWS * HID;
        #pragma unroll
        for (int nt = 0; nt < 4; nt++) {
            int gN = nblk * 64 + nt * 16 + ln;
            float bb = bias[gN];
            #pragma unroll
            for (int r = 0; r < 8; r++) {
                int grow = mblk * 128 + w * 16 + hk * 8 + r;
                dst[(size_t)grow * HID + gN] = acc[mat][nt][r] + bb;
            }
        }
    }
}

// ---------------------------------------------------------------------------
// Kernel 2: one workgroup per (b,i). Structure slice SM[b,i]=[128x64] is read
// once into A fragments; sq/sk/sv computed by WMMA into LDS (bf16). The k_i
// row is copied global->LDS with async-to-LDS ops overlapped with the WMMAs.
// Then scores[h,j] = ((k_i+sk_j).(q_j+sq_j))/8 + mask, relu-normalize, and
// out[h,d] = sum_j p[h,j]*(v[j,h,d]+sv[j,d]).
// ---------------------------------------------------------------------------
__global__ __launch_bounds__(256) void attn_kernel(
    const float* __restrict__ amask, const float* __restrict__ structure,
    const __bf16* __restrict__ wsmall,
    const float* __restrict__ bsq, const float* __restrict__ bsk,
    const float* __restrict__ bsv,
    const float* __restrict__ qkv, float* __restrict__ out) {
    __shared__ __bf16 s_sq[SEQ][68];     // 17408 B each (stride 68 -> 8B aligned)
    __shared__ __bf16 s_sk[SEQ][68];
    __shared__ __bf16 s_sv[SEQ][68];
    __shared__ float  s_kq[NH][HD];      // k row i, all heads     (2 KB)
    __shared__ float  s_sc[NH][SEQ];     // scores -> probs        (4 KB)
    // total LDS = 58368 B

    const int bi = blockIdx.x;
    const int b = bi >> 7, i = bi & 127;
    const int tid  = threadIdx.x;
    const int w    = tid >> 5;
    const int lane = tid & 31;
    const int hk   = lane >> 4;
    const int ln   = lane & 15;

    const float* q = qkv;
    const float* k = qkv + (size_t)ROWS * HID;
    const float* v = qkv + (size_t)2 * ROWS * HID;

    const float* maskrow = amask + (size_t)(b * SEQ + i) * SEQ;
    __builtin_prefetch(maskrow, 0, 1);                 // -> global_prefetch_b8

    // ---- k_i row: async copy global -> LDS (ASYNCcnt path), overlapped with
    //      the A-fragment loads and WMMAs below.
    {
        const float* ksrc = k + (size_t)(b * SEQ + i) * HID;
        unsigned ldsbase = (unsigned)(size_t)(&s_kq[0][0]);
        #pragma unroll
        for (int rep = 0; rep < 2; rep++) {
            unsigned o4 = (unsigned)(tid + rep * 256) * 4u;
            asm volatile("global_load_async_to_lds_b32 %0, %1, %2"
                         :: "v"(ldsbase + o4), "v"(o4), "s"(ksrc)
                         : "memory");
        }
    }

    // ---- A fragments straight from the structure slice (read-once)
    const float* smb = structure + (size_t)(b * SEQ + i) * (SEQ * HD);
    union U16 { v16bf v; __bf16 e[16]; };
    U16 a0, a1;
    {
        const float* arow = smb + (w * 16 + ln) * HD + hk * 8;
        float4 r0 = *(const float4*)(arow + 0);
        float4 r1 = *(const float4*)(arow + 4);
        float4 r2 = *(const float4*)(arow + 16);
        float4 r3 = *(const float4*)(arow + 20);
        float4 r4 = *(const float4*)(arow + 32);
        float4 r5 = *(const float4*)(arow + 36);
        float4 r6 = *(const float4*)(arow + 48);
        float4 r7 = *(const float4*)(arow + 52);
        a0.e[0]  = f2bf(r0.x); a0.e[1]  = f2bf(r0.y); a0.e[2]  = f2bf(r0.z); a0.e[3]  = f2bf(r0.w);
        a0.e[4]  = f2bf(r1.x); a0.e[5]  = f2bf(r1.y); a0.e[6]  = f2bf(r1.z); a0.e[7]  = f2bf(r1.w);
        a0.e[8]  = f2bf(r2.x); a0.e[9]  = f2bf(r2.y); a0.e[10] = f2bf(r2.z); a0.e[11] = f2bf(r2.w);
        a0.e[12] = f2bf(r3.x); a0.e[13] = f2bf(r3.y); a0.e[14] = f2bf(r3.z); a0.e[15] = f2bf(r3.w);
        a1.e[0]  = f2bf(r4.x); a1.e[1]  = f2bf(r4.y); a1.e[2]  = f2bf(r4.z); a1.e[3]  = f2bf(r4.w);
        a1.e[4]  = f2bf(r5.x); a1.e[5]  = f2bf(r5.y); a1.e[6]  = f2bf(r5.z); a1.e[7]  = f2bf(r5.w);
        a1.e[8]  = f2bf(r6.x); a1.e[9]  = f2bf(r6.y); a1.e[10] = f2bf(r6.z); a1.e[11] = f2bf(r6.w);
        a1.e[12] = f2bf(r7.x); a1.e[13] = f2bf(r7.y); a1.e[14] = f2bf(r7.z); a1.e[15] = f2bf(r7.w);
    }

    // ---- sq/sk/sv = SM @ Ws* + bs*  (WMMA, wave w owns rows [16w,16w+16))
    #pragma unroll
    for (int mat = 0; mat < 3; mat++) {
        const __bf16* wb = wsmall + mat * WSMALL_ELEMS;
        const float* bias = (mat == 0) ? bsq : (mat == 1) ? bsk : bsv;
        #pragma unroll
        for (int nt = 0; nt < 4; nt++) {
            v16bf b0 = *(const v16bf*)(wb + (((0 * 4 + nt) * 32) + lane) * 16);
            v16bf b1 = *(const v16bf*)(wb + (((1 * 4 + nt) * 32) + lane) * 16);
            v8f c = {};
            c = wmma_bf16(a0.v, b0, c);
            c = wmma_bf16(a1.v, b1, c);
            int col = nt * 16 + ln;
            float bb = bias[col];
            __bf16 (*dst)[68] = (mat == 0) ? s_sq : (mat == 1) ? s_sk : s_sv;
            #pragma unroll
            for (int r = 0; r < 8; r++) {
                int row = w * 16 + hk * 8 + r;
                dst[row][col] = f2bf(c[r] + bb);
            }
        }
    }
    // async k_i copy must be complete before the barrier
    asm volatile("s_wait_asynccnt 0x0" ::: "memory");
    __syncthreads();

    // ---- scores: 16 lanes per (h,j) pair, coalesced q reads, shfl reduce
    {
        const float maskscale = 0.125f;              // 1/sqrt(64)
        #pragma unroll 2
        for (int it = 0; it < 64; it++) {
            int p = it * 16 + w * 2 + hk;            // 0..1023
            int h = p >> 7, j = p & 127;
            int dl = ln * 4;
            float4 qf = *(const float4*)(q + (size_t)(b * SEQ + j) * HID + h * HD + dl);
            float kk0 = s_kq[h][dl + 0], kk1 = s_kq[h][dl + 1];
            float kk2 = s_kq[h][dl + 2], kk3 = s_kq[h][dl + 3];
            v4bf skv = *(const v4bf*)&s_sk[j][dl];
            v4bf sqv = *(const v4bf*)&s_sq[j][dl];
            float part = (kk0 + bf2f(skv[0])) * (qf.x + bf2f(sqv[0]))
                       + (kk1 + bf2f(skv[1])) * (qf.y + bf2f(sqv[1]))
                       + (kk2 + bf2f(skv[2])) * (qf.z + bf2f(sqv[2]))
                       + (kk3 + bf2f(skv[3])) * (qf.w + bf2f(sqv[3]));
            part += __shfl_xor(part, 1, 32);
            part += __shfl_xor(part, 2, 32);
            part += __shfl_xor(part, 4, 32);
            part += __shfl_xor(part, 8, 32);
            if (ln == 0)
                s_sc[h][j] = part * maskscale + maskrow[j];
        }
    }
    __syncthreads();

    // ---- relu + normalize; wave w handles head row w
    {
        int h = w;
        float r0 = fmaxf(s_sc[h][lane +  0], 0.f);
        float r1 = fmaxf(s_sc[h][lane + 32], 0.f);
        float r2 = fmaxf(s_sc[h][lane + 64], 0.f);
        float r3 = fmaxf(s_sc[h][lane + 96], 0.f);
        float sum = r0 + r1 + r2 + r3;
        sum += __shfl_xor(sum, 1, 32);
        sum += __shfl_xor(sum, 2, 32);
        sum += __shfl_xor(sum, 4, 32);
        sum += __shfl_xor(sum, 8, 32);
        sum += __shfl_xor(sum, 16, 32);
        float inv = 1.f / (sum + 128e-12f);          // sum_j(p + 1e-12)
        s_sc[h][lane +  0] = r0 * inv;
        s_sc[h][lane + 32] = r1 * inv;
        s_sc[h][lane + 64] = r2 * inv;
        s_sc[h][lane + 96] = r3 * inv;
    }
    __syncthreads();

    // ---- context: out[h,d] = sum_j p[h,j] * (v[j,h,d] + sv[j,d])
    {
        float* outb = out + (size_t)(b * SEQ + i) * HID;
        for (int o = tid; o < HID; o += 256) {
            int h = o >> 6, d = o & 63;
            const float* vb = v + (size_t)b * SEQ * HID + h * HD + d;
            float acc = 0.f;
            #pragma unroll 4
            for (int j = 0; j < SEQ; j++)
                acc += s_sc[h][j] * (vb[(size_t)j * HID] + bf2f(s_sv[j][d]));
            outb[o] = acc;
        }
    }
}

// ---------------------------------------------------------------------------
// Launch
// ---------------------------------------------------------------------------
extern "C" void kernel_launch(void* const* d_in, const int* in_sizes, int n_in,
                              void* d_out, int out_size, void* d_ws, size_t ws_size,
                              hipStream_t stream) {
    const float* hidden    = (const float*)d_in[0];
    const float* amask     = (const float*)d_in[1];
    const float* structure = (const float*)d_in[2];
    const float* Wq  = (const float*)d_in[3];  const float* bq  = (const float*)d_in[4];
    const float* Wk  = (const float*)d_in[5];  const float* bk  = (const float*)d_in[6];
    const float* Wv  = (const float*)d_in[7];  const float* bv  = (const float*)d_in[8];
    const float* Wsq = (const float*)d_in[9];  const float* bsq = (const float*)d_in[10];
    const float* Wsk = (const float*)d_in[11]; const float* bsk = (const float*)d_in[12];
    const float* Wsv = (const float*)d_in[13]; const float* bsv = (const float*)d_in[14];
    float* out = (float*)d_out;

    char* ws = (char*)d_ws;
    __bf16* wbig   = (__bf16*)ws;                              // 1,572,864 B
    __bf16* wsmall = (__bf16*)(ws + 3u * WBIG_ELEMS * 2u);     //    24,576 B
    float*  qkv    = (float*)(ws + (2u << 20));                // 50,331,648 B

    // 3*262144 + 3*4096 = 798720 elements == 3120 blocks of 256
    repack_weights<<<3120, 256, 0, stream>>>(Wq, Wk, Wv, Wsq, Wsk, Wsv, wbig, wsmall);
    qkv_kernel<<<dim3(64, 8), 256, 0, stream>>>(hidden, wbig, bq, bk, bv, qkv);
    attn_kernel<<<BATCH * SEQ, 256, 0, stream>>>(amask, structure, wsmall,
                                                 bsq, bsk, bsv, qkv, out);
}
